// GIN_996432413502
// MI455X (gfx1250) — matrix operand order repaired and checked by reading, not compiled, not run on hardware
//
#include <hip/hip_runtime.h>

// Problem constants (match reference)
#define Nn 100000
#define Ee 1600000
#define Hh 128
#define Ll 3
#define Gg 512
#define Cc 10

typedef __attribute__((ext_vector_type(16))) __bf16 v16bf;
typedef __attribute__((ext_vector_type(8)))  float  v8f;
typedef __attribute__((ext_vector_type(4)))  unsigned int v4u;
typedef __attribute__((ext_vector_type(8)))  int  v8i;
typedef __attribute__((ext_vector_type(4)))  int  v4i;

// fp32 -> bf16 round-to-nearest-even
static __device__ __forceinline__ unsigned short f2bf(float f) {
  unsigned int u = __float_as_uint(f);
  unsigned int r = u + 0x7FFFu + ((u >> 16) & 1u);
  return (unsigned short)(r >> 16);
}
static __device__ __forceinline__ unsigned int pack2bf(float a, float b) {
  return (unsigned int)f2bf(a) | ((unsigned int)f2bf(b) << 16);
}

// ---------------------------------------------------------------------------
// TDM: DMA one 128x128 bf16 matrix (row-major, contiguous) from global -> LDS.
// 2D descriptor: data_size=2B, tensor_dim0=tensor_dim1=128, tile covers all,
// dim0 stride = 128 elements. Issued by one wave; EXEC ignored by TDM.
// ---------------------------------------------------------------------------
#if __has_builtin(__builtin_amdgcn_tensor_load_to_lds)
#define HAVE_TDM 1
static __device__ __forceinline__ void tdm_load_w(const unsigned short* gW, unsigned ldsOff) {
  unsigned long long ga = (unsigned long long)(size_t)gW;
  v4u g0;
  g0[0] = 1u;                                   // count=1 user descriptor
  g0[1] = ldsOff;                               // lds_addr (bytes)
  g0[2] = (unsigned)(ga & 0xFFFFFFFFu);         // global_addr[31:0]
  g0[3] = (unsigned)((ga >> 32) & 0x01FFFFFFu)  // global_addr[56:32]
        | (2u << 30);                           // type=2 ("image")
  v8i g1;
  g1[0] = 0x00010000;                           // wg_mask=0, data_size=1 (2B)
  g1[1] = (int)(128u << 16);                    // tensor_dim0[15:0]=128
  g1[2] = (int)(128u << 16);                    // tensor_dim0 hi=0; tensor_dim1[15:0]=128
  g1[3] = (int)(128u << 16);                    // tensor_dim1 hi=0; tile_dim0=128
  g1[4] = 128;                                  // tile_dim1=128; tile_dim2=0
  g1[5] = 128;                                  // tensor_dim0_stride[31:0]=128
  g1[6] = 0;                                    // stride hi=0; dim1_stride (unused for 2D)
  g1[7] = 0;
  v4i z4 = {0, 0, 0, 0};
#if defined(__clang_major__) && (__clang_major__ >= 23)
  v8i z8 = {0, 0, 0, 0, 0, 0, 0, 0};
  __builtin_amdgcn_tensor_load_to_lds(g0, g1, z4, z4, z8, 0);
#else
  __builtin_amdgcn_tensor_load_to_lds(g0, g1, z4, z4, 0);
#endif
}
#else
#define HAVE_TDM 0
#endif

// ---------------------------------------------------------------------------
// agg = feat  (seed for scatter-add; GIN eps=0 so out = segsum + x)
// ---------------------------------------------------------------------------
__global__ void copy_f4(const float4* __restrict__ src, float4* __restrict__ dst, int n4) {
  int i = blockIdx.x * blockDim.x + threadIdx.x;
  if (i < n4) dst[i] = src[i];
}

// ---------------------------------------------------------------------------
// Pre-convert all layer weights fp32 -> bf16 (once per call).
// ---------------------------------------------------------------------------
__global__ void convert_weights(const float* __restrict__ W1s, const float* __restrict__ W2s,
                                unsigned short* __restrict__ w1b, unsigned short* __restrict__ w2b) {
  int i = blockIdx.x * blockDim.x + threadIdx.x;
  if (i < Ll * Hh * Hh) {
    w1b[i] = f2bf(W1s[i]);
    w2b[i] = f2bf(W2s[i]);
  }
}

// ---------------------------------------------------------------------------
// Edge scatter: 32 lanes per edge, float4 gather + 4 atomic f32 adds.
// x[src] table (51 MB) stays L2-resident on MI455X (192 MB L2).
// ---------------------------------------------------------------------------
__global__ void scatter_edges(const float* __restrict__ x,
                              const long long* __restrict__ ei,
                              const float* __restrict__ ew,
                              float* __restrict__ agg) {
  long long tid = (long long)blockIdx.x * blockDim.x + threadIdx.x;
  int e = (int)(tid >> 5);
  int q = (int)(tid & 31);
  if (e >= Ee) return;
  int s = (int)ei[e];
  int d = (int)ei[Ee + e];
  float w = ew[e];
  const float4 v = ((const float4*)(x + (size_t)s * Hh))[q];
  float* a = agg + (size_t)d * Hh + q * 4;
  atomicAdd(a + 0, v.x * w);
  atomicAdd(a + 1, v.y * w);
  atomicAdd(a + 2, v.z * w);
  atomicAdd(a + 3, v.w * w);
}

// ---------------------------------------------------------------------------
// WMMA fragment loads from LDS (contiguous k = 16*(lane/16) + s in both A and B).
// ---------------------------------------------------------------------------
static __device__ __forceinline__ v16bf load_a_frag(const unsigned short* __restrict__ ldsA,
                                                    int rt, int k0, int lane) {
  union { v16bf v; unsigned short u[16]; } f;
  int m = lane & 15;
  int kb = (lane >> 4) * 16;
  const unsigned short* rowp = ldsA + (rt * 16 + m) * Hh;
#pragma unroll
  for (int r = 0; r < 4; ++r) {
    int k = k0 + kb + 2 * r;
    f.u[2 * r]     = rowp[k];
    f.u[2 * r + 1] = rowp[k + 1];
  }
#pragma unroll
  for (int r = 4; r < 8; ++r) {
    int k = k0 + kb + 2 * r;
    f.u[2 * r]     = rowp[k];
    f.u[2 * r + 1] = rowp[k + 1];
  }
  return f.v;
}

static __device__ __forceinline__ v16bf load_b_frag(const unsigned short* __restrict__ ldsW,
                                                    int ct, int k0, int lane) {
  union { v16bf v; unsigned short u[16]; } f;
  int n = lane & 15;
  int kb = (lane >> 4) * 16;
  const unsigned short* rowp = ldsW + (ct * 16 + n) * Hh;
#pragma unroll
  for (int r = 0; r < 8; ++r) {
    int k = k0 + kb + 2 * r;
    f.u[2 * r]     = rowp[k];
    f.u[2 * r + 1] = rowp[k + 1];
  }
  return f.v;
}

// ---------------------------------------------------------------------------
// Fused GIN MLP: out = relu(relu(A @ W1^T + b1) @ W2^T + b2)
// 256 threads = 8 wave32; 64 rows per block; K = 128 = 4 x WMMA bf16 k32.
// Weights arrive via TDM DMA (bf16, pre-converted); LDS 16KB + 32KB.
// ---------------------------------------------------------------------------
__global__ __launch_bounds__(256) void gemm_mlp(
    const float* __restrict__ A,
    const unsigned short* __restrict__ W1b, const float* __restrict__ b1,
    const unsigned short* __restrict__ W2b, const float* __restrict__ b2,
    float* __restrict__ out) {
  __shared__ unsigned short ldsA[64 * Hh];   // activations / hidden, bf16
  __shared__ unsigned short ldsW[Hh * Hh];   // weight, bf16, [j][k]

  int tid  = threadIdx.x;
  int row0 = blockIdx.x * 64;
  unsigned ldsWoff = (unsigned)(size_t)(void*)ldsW;

  // Kick off W1 DMA from one wave, then everyone stages the A tile.
#if HAVE_TDM
  if (tid < 32) tdm_load_w(W1b, ldsWoff);
#else
  {
    const unsigned int* w32 = (const unsigned int*)W1b;
    unsigned int* lw32 = (unsigned int*)ldsW;
    for (int i = tid; i < Hh * Hh / 2; i += 256) lw32[i] = w32[i];
  }
#endif

  // Stage A tile (fp32 -> bf16, packed pairs)
  {
    unsigned int* la32 = (unsigned int*)ldsA;
    for (int i = tid; i < 64 * Hh / 2; i += 256) {
      int r  = i >> 6;        // row (64 u32 per row)
      int k2 = i & 63;        // pair index
      int gr = row0 + r;
      float2 v = make_float2(0.0f, 0.0f);
      if (gr < Nn) v = ((const float2*)(A + (size_t)gr * Hh))[k2];
      la32[i] = pack2bf(v.x, v.y);
    }
  }
#if HAVE_TDM
  if (tid < 32) __builtin_amdgcn_s_wait_tensorcnt(0);
#endif
  __syncthreads();

  int wave = tid >> 5;
  int lane = tid & 31;
  int rt   = wave >> 1;        // row tile 0..3 (16 rows each)
  int ct0  = (wave & 1) * 4;   // col tile base (4 tiles of 16 cols)

  v8f acc[4];
#pragma unroll
  for (int t = 0; t < 4; ++t)
#pragma unroll
    for (int i = 0; i < 8; ++i) acc[t][i] = 0.0f;

  // GEMM1: preload all 4 B frags per k-step, then 4 chained WMMAs.
#pragma unroll
  for (int kk = 0; kk < 4; ++kk) {
    int k0 = kk * 32;
    v16bf af  = load_a_frag(ldsA, rt, k0, lane);
    v16bf bf0 = load_b_frag(ldsW, ct0 + 0, k0, lane);
    v16bf bf1 = load_b_frag(ldsW, ct0 + 1, k0, lane);
    v16bf bf2 = load_b_frag(ldsW, ct0 + 2, k0, lane);
    v16bf bf3 = load_b_frag(ldsW, ct0 + 3, k0, lane);
    acc[0] = __builtin_amdgcn_wmma_f32_16x16x32_bf16(false, af, false, bf0, (short)0, acc[0], false, false);
    acc[1] = __builtin_amdgcn_wmma_f32_16x16x32_bf16(false, af, false, bf1, (short)0, acc[1], false, false);
    acc[2] = __builtin_amdgcn_wmma_f32_16x16x32_bf16(false, af, false, bf2, (short)0, acc[2], false, false);
    acc[3] = __builtin_amdgcn_wmma_f32_16x16x32_bf16(false, af, false, bf3, (short)0, acc[3], false, false);
  }
  __syncthreads();  // all reads of ldsA/ldsW done

  // Start W2 DMA (overlaps the h1 write-back to ldsA).
#if HAVE_TDM
  if (tid < 32) tdm_load_w(W2b, ldsWoff);
#else
  {
    const unsigned int* w32 = (const unsigned int*)W2b;
    unsigned int* lw32 = (unsigned int*)ldsW;
    for (int i = tid; i < Hh * Hh / 2; i += 256) lw32[i] = w32[i];
  }
#endif

  // h1 = relu(acc + b1) -> ldsA (bf16)
  int m0 = (lane >> 4) * 8;   // C/D layout: lanes 16-31 are M+8
  int nn = lane & 15;
#pragma unroll
  for (int t = 0; t < 4; ++t) {
    int col = (ct0 + t) * 16 + nn;
    float bias = b1[col];
#pragma unroll
    for (int r = 0; r < 8; ++r) {
      int row = rt * 16 + m0 + r;
      ldsA[row * Hh + col] = f2bf(fmaxf(acc[t][r] + bias, 0.0f));
    }
  }
#if HAVE_TDM
  if (tid < 32) __builtin_amdgcn_s_wait_tensorcnt(0);
#endif
  __syncthreads();

  // GEMM2
#pragma unroll
  for (int t = 0; t < 4; ++t)
#pragma unroll
    for (int i = 0; i < 8; ++i) acc[t][i] = 0.0f;
#pragma unroll
  for (int kk = 0; kk < 4; ++kk) {
    int k0 = kk * 32;
    v16bf af  = load_a_frag(ldsA, rt, k0, lane);
    v16bf bf0 = load_b_frag(ldsW, ct0 + 0, k0, lane);
    v16bf bf1 = load_b_frag(ldsW, ct0 + 1, k0, lane);
    v16bf bf2 = load_b_frag(ldsW, ct0 + 2, k0, lane);
    v16bf bf3 = load_b_frag(ldsW, ct0 + 3, k0, lane);
    acc[0] = __builtin_amdgcn_wmma_f32_16x16x32_bf16(false, af, false, bf0, (short)0, acc[0], false, false);
    acc[1] = __builtin_amdgcn_wmma_f32_16x16x32_bf16(false, af, false, bf1, (short)0, acc[1], false, false);
    acc[2] = __builtin_amdgcn_wmma_f32_16x16x32_bf16(false, af, false, bf2, (short)0, acc[2], false, false);
    acc[3] = __builtin_amdgcn_wmma_f32_16x16x32_bf16(false, af, false, bf3, (short)0, acc[3], false, false);
  }

  // out = relu(acc + b2), fp32
#pragma unroll
  for (int t = 0; t < 4; ++t) {
    int col = (ct0 + t) * 16 + nn;
    float bias = b2[col];
#pragma unroll
    for (int r = 0; r < 8; ++r) {
      int row = row0 + rt * 16 + m0 + r;
      if (row < Nn) out[(size_t)row * Hh + col] = fmaxf(acc[t][r] + bias, 0.0f);
    }
  }
}

// ---------------------------------------------------------------------------
// Mean pooling + classifier head
// ---------------------------------------------------------------------------
__global__ void zero_pool(float* pooled, float* counts) {
  int i = blockIdx.x * blockDim.x + threadIdx.x;
  if (i < Gg * Hh) pooled[i] = 0.0f;
  if (i < Gg) counts[i] = 0.0f;
}

__global__ void pool_nodes(const float* __restrict__ last,
                           const long long* __restrict__ batch,
                           float* __restrict__ pooled, float* __restrict__ counts) {
  long long tid = (long long)blockIdx.x * blockDim.x + threadIdx.x;
  int n = (int)(tid >> 5);
  int q = (int)(tid & 31);
  if (n >= Nn) return;
  int g = (int)batch[n];
  const float4 v = ((const float4*)(last + (size_t)n * Hh))[q];
  float* p = pooled + (size_t)g * Hh + q * 4;
  atomicAdd(p + 0, v.x);
  atomicAdd(p + 1, v.y);
  atomicAdd(p + 2, v.z);
  atomicAdd(p + 3, v.w);
  if (q == 0) atomicAdd(counts + g, 1.0f);
}

__global__ __launch_bounds__(128) void head(
    const float* __restrict__ pooled, const float* __restrict__ counts,
    const float* __restrict__ lin1_w, const float* __restrict__ lin1_b,
    const float* __restrict__ lin2_w, const float* __restrict__ lin2_b,
    float* __restrict__ out) {
  __shared__ float prow[Hh];
  __shared__ float zrow[Hh];
  __shared__ float logit[Cc];
  int g = blockIdx.x;
  int t = threadIdx.x;
  float cnt = fmaxf(counts[g], 1.0f);
  prow[t] = pooled[(size_t)g * Hh + t] / cnt;
  __syncthreads();
  float s = lin1_b[t];
  for (int k = 0; k < Hh; ++k) s += prow[k] * lin1_w[t * Hh + k];
  zrow[t] = fmaxf(s, 0.0f);
  __syncthreads();
  if (t < Cc) {
    float s2 = lin2_b[t];
    for (int k = 0; k < Hh; ++k) s2 += zrow[k] * lin2_w[t * Hh + k];
    logit[t] = s2;
  }
  __syncthreads();
  if (t == 0) {
    float mx = logit[0];
    for (int c = 1; c < Cc; ++c) mx = fmaxf(mx, logit[c]);
    float se = 0.0f;
    for (int c = 0; c < Cc; ++c) se += expf(logit[c] - mx);
    float lse = mx + logf(se);
    for (int c = 0; c < Cc; ++c) out[g * Cc + c] = logit[c] - lse;
  }
}

// ---------------------------------------------------------------------------
extern "C" void kernel_launch(void* const* d_in, const int* in_sizes, int n_in,
                              void* d_out, int out_size, void* d_ws, size_t ws_size,
                              hipStream_t stream) {
  const float*     x      = (const float*)d_in[0];
  const long long* ei     = (const long long*)d_in[1];   // int64 [2, E]
  const float*     ew     = (const float*)d_in[2];
  const long long* batch  = (const long long*)d_in[3];   // int64 [N], sorted
  const float*     W1s    = (const float*)d_in[4];
  const float*     b1s    = (const float*)d_in[5];
  const float*     W2s    = (const float*)d_in[6];
  const float*     b2s    = (const float*)d_in[7];
  const float*     lin1_w = (const float*)d_in[8];
  const float*     lin1_b = (const float*)d_in[9];
  const float*     lin2_w = (const float*)d_in[10];
  const float*     lin2_b = (const float*)d_in[11];

  float* out        = (float*)d_out;
  float* logits_out = out;            // [G*C]
  float* last_out   = out + Gg * Cc;  // [N*H]

  char*  ws     = (char*)d_ws;
  size_t offB   = (size_t)Nn * Hh * 4;
  size_t offP   = 2 * offB;
  float* bufA   = (float*)ws;                       // agg  [N*H]
  float* bufB   = (float*)(ws + offB);              // hid  [N*H]
  float* pooled = (float*)(ws + offP);              // [G*H]
  float* counts = pooled + Gg * Hh;                 // [G]
  unsigned short* w1b = (unsigned short*)(counts + Gg);     // [L*H*H] bf16
  unsigned short* w2b = w1b + (size_t)Ll * Hh * Hh;         // [L*H*H] bf16

  const int NH4 = Nn * Hh / 4;
  dim3 blk(256);

  convert_weights<<<(Ll * Hh * Hh + 255) / 256, blk, 0, stream>>>(W1s, W2s, w1b, w2b);

  const float* feat = x;
  for (int l = 0; l < Ll; ++l) {
    copy_f4<<<(NH4 + 255) / 256, blk, 0, stream>>>((const float4*)feat, (float4*)bufA, NH4);
    long long sth = (long long)Ee * 32;
    scatter_edges<<<(unsigned)((sth + 255) / 256), blk, 0, stream>>>(feat, ei, ew, bufA);
    float* o = (l == Ll - 1) ? last_out : bufB;
    gemm_mlp<<<(Nn + 63) / 64, blk, 0, stream>>>(bufA,
                                                 w1b + (size_t)l * Hh * Hh, b1s + l * Hh,
                                                 w2b + (size_t)l * Hh * Hh, b2s + l * Hh, o);
    feat = o;
  }

  zero_pool<<<(Gg * Hh + 255) / 256, blk, 0, stream>>>(pooled, counts);
  long long pth = (long long)Nn * 32;
  pool_nodes<<<(unsigned)((pth + 255) / 256), blk, 0, stream>>>(feat, batch, pooled, counts);
  head<<<Gg, dim3(128), 0, stream>>>(pooled, counts, lin1_w, lin1_b, lin2_w, lin2_b, logits_out);
}
